// Model_38585986187698
// MI455X (gfx1250) — matrix operand order repaired and checked by reading, not compiled, not run on hardware
//
#include <hip/hip_runtime.h>
#include <hip/hip_bf16.h>
#include <math.h>

#define V_ 32000
#define E_ 128
#define H_ 128
#define B_ 256
#define T_ 32
#define G4 512   // 4*H

typedef __attribute__((ext_vector_type(16))) __bf16 v16bf;
typedef __attribute__((ext_vector_type(8)))  float  v8f;

// ---------------- WMMA fragment loaders (CDNA5 ISA 7.12.2, wave32) ----------------
// A: 16x32 bf16, row-major source with leading dim ldk.
// lane m=L&15 holds row m; VGPR0-3 = K[half*8 .. half*8+7], VGPR4-7 = K[16+half*8 ..].
__device__ __forceinline__ v16bf load_a_frag(const __bf16* base, int ldk, int kbase, int lane) {
  const int m = lane & 15, half = (lane >> 4) & 1;
  const __bf16* p = base + (size_t)m * ldk + kbase + half * 8;
  v16bf f;
  ((uint4*)&f)[0] = *(const uint4*)(p);        // K = kbase+half*8   .. +7
  ((uint4*)&f)[1] = *(const uint4*)(p + 16);   // K = kbase+16+half*8.. +7
  return f;
}

// B: 32x16 bf16 where B[k][n] = W[colbase+n][k], W row-major [N][ldk].
// lane n=L&15 holds column n; VGPR0-3 = K[half*8 .. half*8+7], VGPR4-7 = K[16+half*8 ..].
__device__ __forceinline__ v16bf load_b_frag(const __bf16* w, int ldk, int colbase,
                                             int kbase, int lane) {
  const int n = lane & 15, half = (lane >> 4) & 1;
  const __bf16* p = w + (size_t)(colbase + n) * ldk + kbase + half * 8;
  v16bf f;
  ((uint4*)&f)[0] = *(const uint4*)(p);
  ((uint4*)&f)[1] = *(const uint4*)(p + 16);
  return f;
}

// ---------------- prep kernels ----------------
__global__ void cvt_f32_bf16(const float* __restrict__ src, __bf16* __restrict__ dst, int n) {
  int i = blockIdx.x * blockDim.x + threadIdx.x;
  if (i < n) dst[i] = (__bf16)src[i];
}

__global__ void embed_gather(const int* __restrict__ seq, const float* __restrict__ emb,
                             __bf16* __restrict__ X) {
  int i = blockIdx.x * blockDim.x + threadIdx.x;   // over T*B*E, row r = t*B+b
  if (i >= T_ * B_ * E_) return;
  int e = i & (E_ - 1);
  int r = i >> 7;
  int t = r >> 8;            // / B_(=256)
  int b = r & (B_ - 1);
  int tok = seq[b * T_ + t];
  X[i] = (__bf16)emb[(size_t)tok * E_ + e];
}

__global__ void bias_sum(const float* __restrict__ a, const float* __restrict__ b,
                         float* __restrict__ o) {
  int i = blockIdx.x * blockDim.x + threadIdx.x;
  if (i < G4) o[i] = a[i] + b[i];
}

// ---------------- K1: Gx[t*B+b, 4H] = X @ w_ih^T + (b_ih+b_hh) ----------------
__global__ void __launch_bounds__(256) gx_gemm(const __bf16* __restrict__ X,
                                               const __bf16* __restrict__ Wih,
                                               const float* __restrict__ bias,
                                               float* __restrict__ Gx) {
  const int lane = threadIdx.x & 31, wave = threadIdx.x >> 5;
  const int ntile = blockIdx.x * 8 + wave;       // 0..31
  const int mrow  = blockIdx.y * 16;             // 0..8176
  const int n = lane & 15, half = (lane >> 4) & 1;
  v8f c;
  const float bv = bias[ntile * 16 + n];
  #pragma unroll
  for (int r = 0; r < 8; ++r) c[r] = bv;
  #pragma unroll
  for (int kk = 0; kk < 4; ++kk) {
    v16bf a = load_a_frag(X + (size_t)mrow * E_, E_, kk * 32, lane);
    v16bf b = load_b_frag(Wih, E_, ntile * 16, kk * 32, lane);
    c = __builtin_amdgcn_wmma_f32_16x16x32_bf16(false, a, false, b, (short)0, c, false, false);
  }
  #pragma unroll
  for (int r = 0; r < 8; ++r)
    Gx[(size_t)(mrow + r + 8 * half) * G4 + ntile * 16 + n] = c[r];
}

// ---------------- K2: serial LSTM recurrence, 16 batch rows per block ----------------
__global__ void __launch_bounds__(256) lstm_recurrence(
    const float* __restrict__ h0, const float* __restrict__ c0,
    const float* __restrict__ Gx, const __bf16* __restrict__ Whh,
    __bf16* __restrict__ Hall, float* __restrict__ hf, float* __restrict__ cf) {
  __shared__ __align__(16) __bf16 sh_hbf[16 * H_];   // h in bf16 (A operand)
  __shared__ float sh_c[16 * H_];                    // c master copy (f32)
  __shared__ float sh_g[16 * G4];                    // gate exchange
  const int tid = threadIdx.x;
  const int lane = tid & 31, wave = tid >> 5;
  const int b0 = blockIdx.x * 16;
  const int n = lane & 15, half = (lane >> 4) & 1;

  for (int i = tid; i < 16 * H_; i += 256) {
    int m = i >> 7, h = i & 127;
    sh_c[i]   = c0[(size_t)(b0 + m) * H_ + h];
    sh_hbf[i] = (__bf16)h0[(size_t)(b0 + m) * H_ + h];
  }
  __syncthreads();

  for (int t = 0; t < T_; ++t) {
    // gates[16,512] = Gx[t,b0:,:] + h @ Whh^T  (each wave: 4 n-tiles, K=128 in 4 steps)
    #pragma unroll
    for (int q = 0; q < 4; ++q) {
      const int nt = wave * 4 + q;
      v8f c;
      #pragma unroll
      for (int r = 0; r < 8; ++r)
        c[r] = Gx[(size_t)(t * B_ + b0 + r + 8 * half) * G4 + nt * 16 + n];
      #pragma unroll
      for (int kk = 0; kk < 4; ++kk) {
        v16bf a = load_a_frag(sh_hbf, H_, kk * 32, lane);
        v16bf b = load_b_frag(Whh, H_, nt * 16, kk * 32, lane);
        c = __builtin_amdgcn_wmma_f32_16x16x32_bf16(false, a, false, b, (short)0, c, false, false);
      }
      #pragma unroll
      for (int r = 0; r < 8; ++r)
        sh_g[(r + 8 * half) * G4 + nt * 16 + n] = c[r];
    }
    __syncthreads();
    // cell update: 16*128 elements, 8 per thread
    #pragma unroll
    for (int q = 0; q < 8; ++q) {
      int i = tid + q * 256;
      int m = i >> 7, h = i & 127;
      float gi = sh_g[m * G4 + h];
      float gf = sh_g[m * G4 + 128 + h];
      float gg = sh_g[m * G4 + 256 + h];
      float go = sh_g[m * G4 + 384 + h];
      gi = 1.f / (1.f + __expf(-gi));
      gf = 1.f / (1.f + __expf(-gf));
      gg = tanhf(gg);
      go = 1.f / (1.f + __expf(-go));
      float cn = gf * sh_c[i] + gi * gg;
      float hn = go * tanhf(cn);
      sh_c[i] = cn;
      sh_hbf[i] = (__bf16)hn;
      Hall[((size_t)(b0 + m) * T_ + t) * H_ + h] = (__bf16)hn;   // row order = b*T+t
      if (t == T_ - 1) {
        hf[(size_t)(b0 + m) * H_ + h] = hn;
        cf[(size_t)(b0 + m) * H_ + h] = cn;
      }
    }
    __syncthreads();
  }
}

// ---------------- K3: out[8192,32000] = Hall @ fc_w^T + fc_b ----------------
// Each wave: one 16-col tile x 4 row tiles (64 rows), reusing B fragments 4x.
__global__ void __launch_bounds__(256) out_gemm(const __bf16* __restrict__ Hall,
                                                const __bf16* __restrict__ Wfc,
                                                const float* __restrict__ fcb,
                                                float* __restrict__ out) {
  const int lane = threadIdx.x & 31, wave = threadIdx.x >> 5;
  const int colbase = (blockIdx.x * 8 + wave) * 16;  // < 32000
  const int mbase   = blockIdx.y * 64;               // < 8192
  const int n = lane & 15, half = (lane >> 4) & 1;

  v16bf bfr[4];
  #pragma unroll
  for (int kk = 0; kk < 4; ++kk)
    bfr[kk] = load_b_frag(Wfc, H_, colbase, kk * 32, lane);

  const float bv = fcb[colbase + n];
  v8f acc[4];
  #pragma unroll
  for (int mm = 0; mm < 4; ++mm)
    #pragma unroll
    for (int r = 0; r < 8; ++r) acc[mm][r] = bv;

  #pragma unroll
  for (int kk = 0; kk < 4; ++kk)
    #pragma unroll
    for (int mm = 0; mm < 4; ++mm) {
      v16bf a = load_a_frag(Hall + (size_t)(mbase + mm * 16) * H_, H_, kk * 32, lane);
      acc[mm] = __builtin_amdgcn_wmma_f32_16x16x32_bf16(false, a, false, bfr[kk],
                                                        (short)0, acc[mm], false, false);
    }

  #pragma unroll
  for (int mm = 0; mm < 4; ++mm)
    #pragma unroll
    for (int r = 0; r < 8; ++r) {
      int row = mbase + mm * 16 + 8 * half + r;
      out[(size_t)row * V_ + colbase + n] = acc[mm][r];
    }
}

// ---------------- host launch ----------------
extern "C" void kernel_launch(void* const* d_in, const int* in_sizes, int n_in,
                              void* d_out, int out_size, void* d_ws, size_t ws_size,
                              hipStream_t stream) {
  const int*   seq = (const int*)d_in[0];
  const float* h0  = (const float*)d_in[1];
  const float* c0  = (const float*)d_in[2];
  const float* emb = (const float*)d_in[3];
  const float* wih = (const float*)d_in[4];
  const float* whh = (const float*)d_in[5];
  const float* bih = (const float*)d_in[6];
  const float* bhh = (const float*)d_in[7];
  const float* fcw = (const float*)d_in[8];
  const float* fcb = (const float*)d_in[9];
  float* out = (float*)d_out;

  char* ws = (char*)d_ws;
  size_t off = 0;
  __bf16* fcw_bf = (__bf16*)(ws + off); off += (size_t)V_ * H_ * 2;   // 8,192,000
  __bf16* wih_bf = (__bf16*)(ws + off); off += (size_t)G4 * H_ * 2;
  __bf16* whh_bf = (__bf16*)(ws + off); off += (size_t)G4 * H_ * 2;
  __bf16* X_bf   = (__bf16*)(ws + off); off += (size_t)T_ * B_ * E_ * 2;
  float*  bias   = (float*) (ws + off); off += (size_t)G4 * 4;
  float*  Gx     = (float*) (ws + off); off += (size_t)T_ * B_ * G4 * 4;
  __bf16* Hall   = (__bf16*)(ws + off); off += (size_t)T_ * B_ * H_ * 2;
  // total ~29.5 MB of workspace

  float* hf = out + (size_t)B_ * T_ * V_;   // tuple order: out, h_f, c_f
  float* cf = hf + (size_t)B_ * H_;

  cvt_f32_bf16<<<(V_ * H_ + 255) / 256, 256, 0, stream>>>(fcw, fcw_bf, V_ * H_);
  cvt_f32_bf16<<<(G4 * H_ + 255) / 256, 256, 0, stream>>>(wih, wih_bf, G4 * H_);
  cvt_f32_bf16<<<(G4 * H_ + 255) / 256, 256, 0, stream>>>(whh, whh_bf, G4 * H_);
  embed_gather<<<(T_ * B_ * E_ + 255) / 256, 256, 0, stream>>>(seq, emb, X_bf);
  bias_sum<<<2, 256, 0, stream>>>(bih, bhh, bias);

  gx_gemm<<<dim3(4, 512), 256, 0, stream>>>(X_bf, wih_bf, bias, Gx);
  lstm_recurrence<<<16, 256, 0, stream>>>(h0, c0, Gx, whh_bf, Hall, hf, cf);
  out_gemm<<<dim3(250, 128), 256, 0, stream>>>(Hall, fcw_bf, fcb, out);
}
